// TransformerGNN_73710228734582
// MI455X (gfx1250) — matrix operand order repaired and checked by reading, not compiled, not run on hardware
//
#include <hip/hip_runtime.h>
#include <hip/hip_bf16.h>

// ---------------------------------------------------------------------------
// TransformerGNN forward for MI455X (gfx1250), wave32 + WMMA f16 (f32 accum).
// Double-buffered LDS staging so global fetch overlaps v_wmma issue.
// ---------------------------------------------------------------------------

typedef _Float16 v16h __attribute__((ext_vector_type(16)));
typedef _Float16 v8h  __attribute__((ext_vector_type(8)));
typedef float    v8f  __attribute__((ext_vector_type(8)));
typedef float    v4f  __attribute__((ext_vector_type(4)));

#define N_NODES   16384
#define N_GRAPHS  64
#define MAX_NODES 512
#define DMODEL    256
#define NHEAD     4
#define HEADD     64
#define FFDIM     2048
#define NROWS     (N_GRAPHS * MAX_NODES)   // 32768 padded rows
#define LN_EPS    1e-5f

__device__ __forceinline__ v8h cvt8(const float* __restrict__ p) {
  v8h r;
#pragma unroll
  for (int i = 0; i < 8; ++i) r[i] = (_Float16)p[i];
  return r;
}
__device__ __forceinline__ v16h cat16(v8h lo, v8h hi) {
  return __builtin_shufflevector(lo, hi, 0,1,2,3,4,5,6,7,8,9,10,11,12,13,14,15);
}
__device__ __forceinline__ v8f wmma16(v16h a, v16h b, v8f c) {
  return __builtin_amdgcn_wmma_f32_16x16x32_f16(false, a, false, b, (short)0, c,
                                                false, false);
}
__device__ __forceinline__ v8f vzero8() {
  v8f z;
#pragma unroll
  for (int i = 0; i < 8; ++i) z[i] = 0.0f;
  return z;
}

// ---------------------------------------------------------------------------
// Generic WMMA GEMM:  C[m,n] = act( sum_k A[m,k] * W[n,k] + bias[n] )
// Block tile 128(M) x 64(N), 8 waves, each wave 32x32 (2x2 WMMA frags).
// K staged through double-buffered LDS in f16 chunks of 32
// (row stride padded to 40 halves to spread banks).
// ---------------------------------------------------------------------------
template <typename AT, bool RELU, bool OUTF16, bool REMAP>
__global__ __launch_bounds__(256) void gemm_wmma_k(
    const AT* __restrict__ A,            // [M, K] row-major
    const float* __restrict__ W,         // [N, K] row-major
    const float* __restrict__ bias,      // [N]
    void* __restrict__ Cout, int ldc,    // [M(out), ldc]
    const int* __restrict__ remap,       // optional output-row remap
    int K) {
  __shared__ _Float16 As[2][128 * 40];
  __shared__ _Float16 Bs[2][64 * 40];

  const int tid  = threadIdx.x;
  const int bm   = blockIdx.x * 128;
  const int bn   = blockIdx.y * 64;
  const int w    = tid >> 5;
  const int lane = tid & 31;
  const int wm   = (w & 3) * 32;
  const int wn   = (w >> 2) * 32;
  const int ml    = lane & 15;
  const int khalf = lane >> 4;

  const int arow = tid >> 1;           // 0..127
  const int acol = (tid & 1) * 16;     // 0 / 16
  const int brow = tid >> 2;           // 0..63
  const int bcol = (tid & 3) * 8;      // 0,8,16,24

  const AT*    aptr = A + (size_t)(bm + arow) * K + acol;
  const float* bptr = W + (size_t)(bn + brow) * K + bcol;
  _Float16* adst = nullptr;  // set per-buffer below
  (void)adst;

  v8f acc[2][2];
#pragma unroll
  for (int i = 0; i < 2; ++i)
#pragma unroll
    for (int j = 0; j < 2; ++j) acc[i][j] = vzero8();

  // prologue: stage chunk 0 into buffer 0
  {
    _Float16* da = &As[0][arow * 40 + acol];
    _Float16* db = &Bs[0][brow * 40 + bcol];
#pragma unroll
    for (int i = 0; i < 16; ++i) da[i] = (_Float16)aptr[i];
#pragma unroll
    for (int i = 0; i < 8; ++i)  db[i] = (_Float16)bptr[i];
  }
  __syncthreads();

  const int nk = K >> 5;
  for (int kc = 0; kc < nk; ++kc) {
    const int cur = kc & 1;
    // prefetch next chunk into registers (overlaps with WMMA below)
    AT    areg[16];
    float breg[8];
    const bool more = (kc + 1) < nk;
    if (more) {
      const AT*    ap = aptr + (kc + 1) * 32;
      const float* bp = bptr + (kc + 1) * 32;
#pragma unroll
      for (int i = 0; i < 16; ++i) areg[i] = ap[i];
#pragma unroll
      for (int i = 0; i < 8; ++i)  breg[i] = bp[i];
    }

    // compute from current buffer
    v16h af[2], bf[2];
#pragma unroll
    for (int ms = 0; ms < 2; ++ms) {
      const _Float16* p = &As[cur][(wm + ms * 16 + ml) * 40 + khalf * 8];
      af[ms] = cat16(*(const v8h*)p, *(const v8h*)(p + 16));
    }
#pragma unroll
    for (int ns = 0; ns < 2; ++ns) {
      const _Float16* p = &Bs[cur][(wn + ns * 16 + ml) * 40 + khalf * 8];
      bf[ns] = cat16(*(const v8h*)p, *(const v8h*)(p + 16));
    }
#pragma unroll
    for (int ms = 0; ms < 2; ++ms)
#pragma unroll
      for (int ns = 0; ns < 2; ++ns)
        acc[ms][ns] = wmma16(af[ms], bf[ns], acc[ms][ns]);

    // store prefetched chunk into alternate buffer
    if (more) {
      const int nxt = cur ^ 1;
      _Float16* da = &As[nxt][arow * 40 + acol];
      _Float16* db = &Bs[nxt][brow * 40 + bcol];
#pragma unroll
      for (int i = 0; i < 16; ++i) da[i] = (_Float16)areg[i];
#pragma unroll
      for (int i = 0; i < 8; ++i)  db[i] = (_Float16)breg[i];
    }
    __syncthreads();
  }

  const int rh = khalf * 8;
#pragma unroll
  for (int ms = 0; ms < 2; ++ms) {
#pragma unroll
    for (int ns = 0; ns < 2; ++ns) {
      const int n  = bn + wn + ns * 16 + ml;
      const float bv = bias[n];
#pragma unroll
      for (int r = 0; r < 8; ++r) {
        const int row = bm + wm + ms * 16 + rh + r;
        float v = acc[ms][ns][r] + bv;
        if constexpr (RELU) v = fmaxf(v, 0.0f);
        long orow = row;
        if constexpr (REMAP) {
          int rm = remap[row];
          if (rm < 0) continue;
          orow = rm;
        }
        if constexpr (OUTF16)
          ((_Float16*)Cout)[(size_t)orow * ldc + n] = (_Float16)v;
        else
          ((float*)Cout)[(size_t)orow * ldc + n] = v;
      }
    }
  }
}

// ---------------------------------------------------------------------------
// Attention scores: S[g,h,q,k] = mask(scale * Q.Kt), stored f16.
// Block = 128 thr (4 waves); block covers 64 q rows of one (g, head).
// ---------------------------------------------------------------------------
__global__ __launch_bounds__(128) void attn_scores_k(
    const float* __restrict__ qkv,       // [NROWS, 768]
    const int* __restrict__ counts,      // [N_GRAPHS]
    _Float16* __restrict__ S) {          // [N_GRAPHS*NHEAD, 512, 512]
  const int qt = blockIdx.x, hd = blockIdx.y, g = blockIdx.z;
  const int w = threadIdx.x >> 5, lane = threadIdx.x & 31;
  const int ml = lane & 15, khalf = lane >> 4;
  const int qbase = qt * 64 + w * 16;

  const float* qrow = qkv + ((size_t)g * 512 + qbase + ml) * 768 + hd * 64;
  v16h qf[2];
#pragma unroll
  for (int kc = 0; kc < 2; ++kc) {
    const float* p = qrow + kc * 32 + khalf * 8;
    qf[kc] = cat16(cvt8(p), cvt8(p + 16));
  }

  const int cnt = counts[g];
  const float scale = 0.125f;  // 1/sqrt(64)
  const float* kbase = qkv + (size_t)g * 512 * 768 + 256 + hd * 64;
  _Float16* srow = S + (size_t)(g * NHEAD + hd) * 512 * 512;

  for (int cg = 0; cg < 8; ++cg) {
    v8f acc[4];
#pragma unroll
    for (int i = 0; i < 4; ++i) acc[i] = vzero8();
#pragma unroll
    for (int ns = 0; ns < 4; ++ns) {
      const int ncol = cg * 64 + ns * 16 + ml;
      const float* kp = kbase + (size_t)ncol * 768 + khalf * 8;
#pragma unroll
      for (int kc = 0; kc < 2; ++kc) {
        const float* p = kp + kc * 32;
        v16h bf = cat16(cvt8(p), cvt8(p + 16));
        acc[ns] = wmma16(qf[kc], bf, acc[ns]);
      }
    }
#pragma unroll
    for (int ns = 0; ns < 4; ++ns) {
      const int ncol = cg * 64 + ns * 16 + ml;
      const bool valid = ncol < cnt;
#pragma unroll
      for (int r = 0; r < 8; ++r) {
        const int qr = qbase + khalf * 8 + r;
        const float v = valid ? acc[ns][r] * scale : -30000.0f;
        srow[(size_t)qr * 512 + ncol] = (_Float16)v;
      }
    }
  }
}

// ---------------------------------------------------------------------------
// Row softmax over 512 f16 scores, in place.  One wave per row.
// ---------------------------------------------------------------------------
__global__ __launch_bounds__(256) void softmax_rows_k(_Float16* __restrict__ S) {
  const int w = threadIdx.x >> 5, lane = threadIdx.x & 31;
  const size_t row = (size_t)blockIdx.x * 8 + w;
  _Float16* p = S + row * 512 + lane * 16;
  float v[16];
  {
    v8h a = *(const v8h*)p;
    v8h b = *(const v8h*)(p + 8);
#pragma unroll
    for (int i = 0; i < 8; ++i) { v[i] = (float)a[i]; v[8 + i] = (float)b[i]; }
  }
  float mx = v[0];
#pragma unroll
  for (int i = 1; i < 16; ++i) mx = fmaxf(mx, v[i]);
  for (int off = 16; off > 0; off >>= 1) mx = fmaxf(mx, __shfl_xor(mx, off));
  float sum = 0.0f;
#pragma unroll
  for (int i = 0; i < 16; ++i) { v[i] = expf(v[i] - mx); sum += v[i]; }
  for (int off = 16; off > 0; off >>= 1) sum += __shfl_xor(sum, off);
  const float inv = 1.0f / sum;
  v8h a, b;
#pragma unroll
  for (int i = 0; i < 8; ++i) {
    a[i] = (_Float16)(v[i] * inv);
    b[i] = (_Float16)(v[8 + i] * inv);
  }
  *(v8h*)p = a;
  *(v8h*)(p + 8) = b;
}

// ---------------------------------------------------------------------------
// ctx[g,q,hd*64+c] = P @ V.  V chunks transposed into double-buffered LDS.
// Block = 128 thr (4 waves); covers 64 q rows of one (g, head).
// ---------------------------------------------------------------------------
__global__ __launch_bounds__(128) void attn_ctx_k(
    const float* __restrict__ qkv,       // [NROWS, 768]
    const _Float16* __restrict__ P,      // softmaxed scores
    float* __restrict__ ctx) {           // [NROWS, 256]
  __shared__ _Float16 Vt[2][64 * 40];    // V^T chunk: [col][k], stride 40
  const int qt = blockIdx.x, hd = blockIdx.y, g = blockIdx.z;
  const int w = threadIdx.x >> 5, lane = threadIdx.x & 31;
  const int ml = lane & 15, khalf = lane >> 4;
  const int qbase = qt * 64 + w * 16;

  const _Float16* prow = P + ((size_t)(g * NHEAD + hd) * 512 + qbase + ml) * 512;
  const int skr = threadIdx.x >> 2;        // 0..31
  const int sc0 = (threadIdx.x & 3) * 16;  // 0,16,32,48
  const float* vbase = qkv + ((size_t)g * 512 + skr) * 768 + 512 + hd * 64 + sc0;

  v8f acc[4];
#pragma unroll
  for (int i = 0; i < 4; ++i) acc[i] = vzero8();

  // prologue: stage chunk 0
  {
#pragma unroll
    for (int i = 0; i < 16; ++i) Vt[0][(sc0 + i) * 40 + skr] = (_Float16)vbase[i];
  }
  __syncthreads();

  for (int kc = 0; kc < 512; kc += 32) {
    const int cur = (kc >> 5) & 1;
    const bool more = (kc + 32) < 512;
    float vreg[16];
    if (more) {
      const float* vp = vbase + (size_t)(kc + 32) * 768;
#pragma unroll
      for (int i = 0; i < 16; ++i) vreg[i] = vp[i];
    }

    const _Float16* pp = prow + kc + khalf * 8;
    v16h af = cat16(*(const v8h*)pp, *(const v8h*)(pp + 16));
#pragma unroll
    for (int ns = 0; ns < 4; ++ns) {
      const _Float16* bp = &Vt[cur][(ns * 16 + ml) * 40 + khalf * 8];
      v16h bf = cat16(*(const v8h*)bp, *(const v8h*)(bp + 16));
      acc[ns] = wmma16(af, bf, acc[ns]);
    }

    if (more) {
      const int nxt = cur ^ 1;
#pragma unroll
      for (int i = 0; i < 16; ++i) Vt[nxt][(sc0 + i) * 40 + skr] = (_Float16)vreg[i];
    }
    __syncthreads();
  }

#pragma unroll
  for (int ns = 0; ns < 4; ++ns) {
#pragma unroll
    for (int r = 0; r < 8; ++r) {
      const int qr = qbase + khalf * 8 + r;
      ctx[((size_t)g * 512 + qr) * 256 + hd * 64 + ns * 16 + ml] = acc[ns][r];
    }
  }
}

// ---------------------------------------------------------------------------
// h = LayerNorm(h + t) * g + b, per 256-wide row.  One wave per row.
// ---------------------------------------------------------------------------
__global__ __launch_bounds__(256) void residual_ln_k(
    float* __restrict__ h, const float* __restrict__ t,
    const float* __restrict__ gam, const float* __restrict__ bet) {
  const int w = threadIdx.x >> 5, lane = threadIdx.x & 31;
  const size_t row = (size_t)blockIdx.x * 8 + w;
  float* hp = h + row * 256 + lane * 8;
  const float* tp = t + row * 256 + lane * 8;
  float x[8];
  float s = 0.0f, sq = 0.0f;
#pragma unroll
  for (int i = 0; i < 8; ++i) {
    x[i] = hp[i] + tp[i];
    s += x[i];
    sq += x[i] * x[i];
  }
  for (int off = 16; off > 0; off >>= 1) {
    s  += __shfl_xor(s, off);
    sq += __shfl_xor(sq, off);
  }
  const float mean = s * (1.0f / 256.0f);
  const float var  = sq * (1.0f / 256.0f) - mean * mean;
  const float inv  = rsqrtf(var + LN_EPS);
#pragma unroll
  for (int i = 0; i < 8; ++i) {
    const int c = lane * 8 + i;
    hp[i] = (x[i] - mean) * inv * gam[c] + bet[c];
  }
}

// ---------------------------------------------------------------------------
// Mean-pool over 512 rows (incl. pads) + classifier [10,256].
// ---------------------------------------------------------------------------
__global__ __launch_bounds__(256) void pool_cls_k(
    const float* __restrict__ h, const float* __restrict__ Wc,
    const float* __restrict__ bc, float* __restrict__ out) {
  __shared__ float pooled[256];
  const int g = blockIdx.x, d = threadIdx.x;
  float s = 0.0f;
  for (int i = 0; i < 512; ++i) s += h[((size_t)g * 512 + i) * 256 + d];
  pooled[d] = s * (1.0f / 512.0f);
  __syncthreads();
  if (d < 10) {
    float a = bc[d];
    for (int k = 0; k < 256; ++k) a += pooled[k] * Wc[d * 256 + k];
    out[g * 10 + d] = a;
  }
}

// ---------------------------------------------------------------------------
// Ragged bookkeeping + zero-fill.
// ---------------------------------------------------------------------------
__global__ void compute_offsets_k(const int* __restrict__ batch,
                                  int* __restrict__ offs, int* __restrict__ counts) {
  const int g = threadIdx.x;
  int lo = 0, hi = N_NODES;
  while (lo < hi) { int mid = (lo + hi) >> 1; if (batch[mid] < g) lo = mid + 1; else hi = mid; }
  const int lb = lo;
  lo = 0; hi = N_NODES;
  while (lo < hi) { int mid = (lo + hi) >> 1; if (batch[mid] < g + 1) lo = mid + 1; else hi = mid; }
  offs[g] = lb;
  counts[g] = lo - lb;
}

__global__ void compute_rowmap_k(const int* __restrict__ batch,
                                 const int* __restrict__ offs,
                                 int* __restrict__ rowmap) {
  const int i = blockIdx.x * 256 + threadIdx.x;
  const int g = batch[i];
  const int pos = i - offs[g];
  rowmap[i] = (pos < MAX_NODES) ? g * MAX_NODES + pos : -1;
}

__global__ void zero_f32_k(v4f* __restrict__ p) {
  const size_t i = (size_t)blockIdx.x * blockDim.x + threadIdx.x;
  v4f z;
#pragma unroll
  for (int j = 0; j < 4; ++j) z[j] = 0.0f;
  p[i] = z;
}

// ---------------------------------------------------------------------------
extern "C" void kernel_launch(void* const* d_in, const int* in_sizes, int n_in,
                              void* d_out, int out_size, void* d_ws, size_t ws_size,
                              hipStream_t stream) {
  const float* x         = (const float*)d_in[0];
  const int*   batch     = (const int*)  d_in[2];
  const float* W_emb     = (const float*)d_in[3];
  const float* b_emb     = (const float*)d_in[4];
  const float* in_proj_w = (const float*)d_in[5];
  const float* in_proj_b = (const float*)d_in[6];
  const float* out_w     = (const float*)d_in[7];
  const float* out_b     = (const float*)d_in[8];
  const float* ln1_g     = (const float*)d_in[9];
  const float* ln1_b     = (const float*)d_in[10];
  const float* ff1_w     = (const float*)d_in[11];
  const float* ff1_b     = (const float*)d_in[12];
  const float* ff2_w     = (const float*)d_in[13];
  const float* ff2_b     = (const float*)d_in[14];
  const float* ln2_g     = (const float*)d_in[15];
  const float* ln2_b     = (const float*)d_in[16];
  const float* W_cls     = (const float*)d_in[17];
  const float* b_cls     = (const float*)d_in[18];
  float* out = (float*)d_out;

  char* wp = (char*)d_ws;
  auto bump = [&](size_t bytes) {
    void* p = (void*)wp;
    wp += (bytes + 255) & ~(size_t)255;
    return p;
  };
  int*      offs   = (int*)bump(N_GRAPHS * 4);
  int*      counts = (int*)bump(N_GRAPHS * 4);
  int*      rowmap = (int*)bump(N_NODES * 4);
  float*    h      = (float*)bump((size_t)NROWS * DMODEL * 4);
  float*    tmp    = (float*)bump((size_t)NROWS * DMODEL * 4);
  float*    ctxb   = (float*)bump((size_t)NROWS * DMODEL * 4);
  float*    qkvb   = (float*)bump((size_t)NROWS * 3 * DMODEL * 4);
  _Float16* big    = (_Float16*)bump((size_t)NROWS * FFDIM * 2);  // scores & ffmid share

  compute_offsets_k<<<1, 64, 0, stream>>>(batch, offs, counts);
  compute_rowmap_k<<<N_NODES / 256, 256, 0, stream>>>(batch, offs, rowmap);
  zero_f32_k<<<(NROWS * DMODEL) / (256 * 4), 256, 0, stream>>>((v4f*)h);

  // embed: [16384,256] x [256,256]^T -> scattered into padded h
  gemm_wmma_k<float, false, false, true>
      <<<dim3(N_NODES / 128, DMODEL / 64), 256, 0, stream>>>(
          x, W_emb, b_emb, h, DMODEL, rowmap, DMODEL);

  for (int l = 0; l < 2; ++l) {
    // qkv = h @ in_proj^T + b   [32768, 768]
    gemm_wmma_k<float, false, false, false>
        <<<dim3(NROWS / 128, (3 * DMODEL) / 64), 256, 0, stream>>>(
            h, in_proj_w + (size_t)l * 3 * DMODEL * DMODEL,
            in_proj_b + (size_t)l * 3 * DMODEL, qkvb, 3 * DMODEL, nullptr, DMODEL);

    attn_scores_k<<<dim3(8, NHEAD, N_GRAPHS), 128, 0, stream>>>(qkvb, counts, big);
    softmax_rows_k<<<(N_GRAPHS * NHEAD * 512) / 8, 256, 0, stream>>>(big);
    attn_ctx_k<<<dim3(8, NHEAD, N_GRAPHS), 128, 0, stream>>>(qkvb, big, ctxb);

    // out projection -> tmp
    gemm_wmma_k<float, false, false, false>
        <<<dim3(NROWS / 128, DMODEL / 64), 256, 0, stream>>>(
            ctxb, out_w + (size_t)l * DMODEL * DMODEL,
            out_b + (size_t)l * DMODEL, tmp, DMODEL, nullptr, DMODEL);
    residual_ln_k<<<NROWS / 8, 256, 0, stream>>>(h, tmp, ln1_g + l * DMODEL,
                                                 ln1_b + l * DMODEL);

    // ff1 (relu, f16 out) -> big ; ff2 (f16 A, K=2048) -> tmp
    gemm_wmma_k<float, true, true, false>
        <<<dim3(NROWS / 128, FFDIM / 64), 256, 0, stream>>>(
            h, ff1_w + (size_t)l * FFDIM * DMODEL,
            ff1_b + (size_t)l * FFDIM, big, FFDIM, nullptr, DMODEL);
    gemm_wmma_k<_Float16, false, false, false>
        <<<dim3(NROWS / 128, DMODEL / 64), 256, 0, stream>>>(
            big, ff2_w + (size_t)l * DMODEL * FFDIM,
            ff2_b + (size_t)l * DMODEL, tmp, DMODEL, nullptr, FFDIM);
    residual_ln_k<<<NROWS / 8, 256, 0, stream>>>(h, tmp, ln2_g + l * DMODEL,
                                                 ln2_b + l * DMODEL);
  }

  pool_cls_k<<<N_GRAPHS, 256, 0, stream>>>(h, W_cls, b_cls, out);
  (void)in_sizes; (void)n_in; (void)out_size; (void)ws_size;
}
